// adapt_mnist_net_60979945669246
// MI455X (gfx1250) — matrix operand order repaired and checked by reading, not compiled, not run on hardware
//
#include <hip/hip_runtime.h>

typedef __attribute__((ext_vector_type(16))) __bf16 v16bf;
typedef __attribute__((ext_vector_type(8)))  float  v8f;

#define NWAVES   8
#define NTHREADS 256
#define TILE     16   // images per workgroup = M dimension of every WMMA

// ---------------- compile-time for ----------------
template <int I> struct IC { static constexpr int v = I; };
template <int... Is> struct ISeq {};
template <int N, int... Is> struct MkSeq : MkSeq<N - 1, N - 1, Is...> {};
template <int... Is> struct MkSeq<0, Is...> { using T = ISeq<Is...>; };
template <class F, int... Is>
static __device__ __forceinline__ void sfor_impl(F&& f, ISeq<Is...>) { (f(IC<Is>{}), ...); }
template <int N, class F>
static __device__ __forceinline__ void sfor(F&& f) { sfor_impl(f, typename MkSeq<N>::T{}); }

// ---------------- LDS layout (bf16 element offsets) ----------------
// Halo-padded rotating activation regions ([img][C][H+2p][W+2p], halos hold 0):
//   R0 (49152): act1 [16][3][32][32]  (halo2) , later actP2 [16][16][9][9] (halo1)
//   R1 (32768): input [16][1][34][34] (halo3) , later actP1 [16][8][16][16] (halo1),
//               actP3 [16][24][3][3] (halo0)
//   R2 (40960): act3 [16][10][16][16] (halo1)
static constexpr int R0_OFF = 0;
static constexpr int R1_OFF = 49152;
static constexpr int R2_OFF = R1_OFF + 32768;
static constexpr int W1_OFF = R2_OFF + 40960;      //  64*16 (K=49  pad 64,  N=16)
static constexpr int W2_OFF = W1_OFF + 64 * 16;    //  96*16 (K=75  pad 96,  N=16)
static constexpr int W3_OFF = W2_OFF + 96 * 16;    //  96*16 (K=72  pad 96,  N=16)
static constexpr int W4_OFF = W3_OFF + 96 * 16;    //  96*16 (K=90  pad 96,  N=16)
static constexpr int W5_OFF = W4_OFF + 96 * 16;    // 160*32 (K=144 pad 160, N=32)
static constexpr int WD_OFF = W5_OFF + 160 * 32;   // 224*16 (K=216 pad 224, N=16)
static constexpr int LDS_ELTS = WD_OFF + 224 * 16; // 137,216 bf16 = 274,432 B < 320 KB

static __device__ __forceinline__ v8f wmma_bf16(v16bf a, v16bf b, v8f c) {
  return __builtin_amdgcn_wmma_f32_16x16x32_bf16(false, a, false, b, (short)0, c,
                                                 false, false);
}
static __device__ __forceinline__ v8f vzero8() {
  v8f z = {0.f, 0.f, 0.f, 0.f, 0.f, 0.f, 0.f, 0.f};
  return z;
}

// constexpr im2col element offset inside one image (halo coords, Hp=H+2p)
template <int Hp, int Wp, int KH, int KW>
static constexpr int im2off(int k) {
  return (k / (KH * KW)) * Hp * Wp + ((k % (KH * KW)) / KW) * Wp + (k % KW);
}

// ---- A fragment (16x32 bf16). lane&15 = image M; g1 = lane>=16 (K-half select).
// base = m*C*Hp*Wp + h*Wp + w   (h,w = un-padded output pixel; halo absorbs conv PAD)
template <int Hp, int Wp, int KH, int KW, int KTRUE, int K0>
static __device__ __forceinline__ v16bf make_a_frag(const __bf16* act, int base, bool g1) {
  v16bf a;
  sfor<16>([&](auto ii) {
    constexpr int idx = decltype(ii)::v;
    constexpr int v = idx >> 1, p = idx & 1;
    constexpr int kA = K0 + (v & 3) * 2 + (v >> 2) * 16 + p;  // k for lanes 0-15
    constexpr int kB = kA + 8;                                // k for lanes 16-31
    constexpr bool vA = kA < KTRUE, vB = kB < KTRUE;
    if constexpr (!vA && !vB) {
      a[idx] = (__bf16)0.0f;
    } else {
      constexpr int oA = im2off<Hp, Wp, KH, KW>(vA ? kA : 0);
      constexpr int oB = im2off<Hp, Wp, KH, KW>(vB ? kB : 0);
      const int off = g1 ? oB : oA;          // cndmask of two constants (loop-invariant)
      const __bf16 t = act[base + off];
      if constexpr (vA && vB) a[idx] = t;
      else                    a[idx] = ((g1 ? vB : vA)) ? t : (__bf16)0.0f;
    }
  });
  return a;
}

// ---- B fragment (32x16 bf16): all-immediate offsets; baseB = ncol (+8*NSTR if g1).
template <int NSTR, int K0>
static __device__ __forceinline__ v16bf make_b_frag(const __bf16* w, int baseB) {
  v16bf b;
  sfor<16>([&](auto ii) {
    constexpr int idx = decltype(ii)::v;
    constexpr int v = idx >> 1, p = idx & 1;
    constexpr int off = (K0 + (v & 3) * 2 + (v >> 2) * 16 + p) * NSTR;
    b[idx] = w[baseB + off];
  });
  return b;
}

// ---- conv stride1 (same-size) + ReLU; in/out both halo-padded in LDS.
template <int C, int H, int W, int KH, int KW, int PAD, int OCH, int KPAD, int OHALO>
static __device__ __forceinline__ void conv_relu(const __bf16* in, __bf16* out,
                                                 const __bf16* wl, int wave, int lane) {
  constexpr int KTRUE = C * KH * KW, NFRAG = KPAD / 32;
  constexpr int Hp = H + 2 * PAD, Wp = W + 2 * PAD;
  constexpr int HpO = H + 2 * OHALO, WpO = W + 2 * OHALO;
  const int nlo = lane & 15;
  const bool g1 = (lane >> 4) != 0;
  const int mbase = nlo * (C * Hp * Wp);

  v16bf bfr[NFRAG];
  const int baseB = nlo + (g1 ? 8 * 16 : 0);
  sfor<NFRAG>([&](auto fi) {
    constexpr int F = decltype(fi)::v;
    bfr[F] = make_b_frag<16, F * 32>(wl, baseB);
  });

  for (int pos = wave; pos < H * W; pos += NWAVES) {   // wave is SGPR => scalar loop
    const int oh = pos / W, ow = pos % W;
    const int base = mbase + oh * Wp + ow;
    v8f acc = vzero8();
    sfor<NFRAG>([&](auto fi) {
      constexpr int F = decltype(fi)::v;
      v16bf a = make_a_frag<Hp, Wp, KH, KW, KTRUE, F * 32>(in, base, g1);
      acc = wmma_bf16(a, bfr[F], acc);
    });
    sfor<8>([&](auto ri) {
      constexpr int r = decltype(ri)::v;
      const int m = r + (g1 ? 8 : 0);
      if (nlo < OCH)
        out[((m * OCH + nlo) * HpO + oh + OHALO) * WpO + ow + OHALO] =
            (__bf16)fmaxf(acc[r], 0.0f);
    });
  }
}

// ---- conv + ReLU fused with 3x3 stride-2 maxpool; output halo-padded.
template <int C, int H, int W, int KH, int KW, int PAD, int OCH, int KPAD,
          int PH, int PW, int PPAD, int OHALO, int NT>
static __device__ __forceinline__ void conv_relu_pool(const __bf16* in, __bf16* out,
                                                      const __bf16* wl, int wave, int lane) {
  constexpr int KTRUE = C * KH * KW, NFRAG = KPAD / 32, NSTR = NT * 16;
  constexpr int Hp = H + 2 * PAD, Wp = W + 2 * PAD;
  constexpr int HpO = PH + 2 * OHALO, WpO = PW + 2 * OHALO;
  const int nlo = lane & 15;
  const bool g1 = (lane >> 4) != 0;
  const int mbase = nlo * (C * Hp * Wp);

  v16bf bfr[NT][NFRAG];
  const int baseB = nlo + (g1 ? 8 * NSTR : 0);
  sfor<NT>([&](auto ti) {
    constexpr int T = decltype(ti)::v;
    sfor<NFRAG>([&](auto fi) {
      constexpr int F = decltype(fi)::v;
      bfr[T][F] = make_b_frag<NSTR, F * 32>(wl, baseB + T * 16);
    });
  });

  for (int pos = wave; pos < PH * PW; pos += NWAVES) {  // scalar loop
    const int ph = pos / PW, pw = pos % PW;
    v8f best[NT];
    sfor<NT>([&](auto ti) { best[decltype(ti)::v] = vzero8(); });  // relu => max >= 0

    for (int dh = 0; dh < 3; ++dh) {
      for (int dw = 0; dw < 3; ++dw) {
        const int h = ph * 2 - PPAD + dh;   // scalar
        const int w = pw * 2 - PPAD + dw;
        if (h >= 0 && h < H && w >= 0 && w < W) {   // scalar branch
          const int base = mbase + h * Wp + w;
          v16bf afr[NFRAG];
          sfor<NFRAG>([&](auto fi) {
            constexpr int F = decltype(fi)::v;
            afr[F] = make_a_frag<Hp, Wp, KH, KW, KTRUE, F * 32>(in, base, g1);
          });
          sfor<NT>([&](auto ti) {
            constexpr int T = decltype(ti)::v;
            v8f acc = vzero8();
            sfor<NFRAG>([&](auto fi) {
              constexpr int F = decltype(fi)::v;
              acc = wmma_bf16(afr[F], bfr[T][F], acc);
            });
            sfor<8>([&](auto ri) {
              constexpr int r = decltype(ri)::v;
              best[T][r] = fmaxf(best[T][r], acc[r]);
            });
          });
        }
      }
    }
    sfor<NT>([&](auto ti) {
      constexpr int T = decltype(ti)::v;
      sfor<8>([&](auto ri) {
        constexpr int r = decltype(ri)::v;
        const int m = r + (g1 ? 8 : 0), n = T * 16 + nlo;
        if (n < OCH)
          out[((m * OCH + n) * HpO + ph + OHALO) * WpO + pw + OHALO] = (__bf16)best[T][r];
      });
    });
  }
}

// ---- dense: [16 x 216] @ wd^T [216 x 10] -> f32 global (wave 0 only).
static __device__ __forceinline__ void dense_out(const __bf16* x, const __bf16* wl,
                                                 float* out, int imgbase, int lane) {
  const int nlo = lane & 15;
  const bool g1 = (lane >> 4) != 0;
  const int baseA = nlo * 216 + (g1 ? 8 : 0);       // grp K-shift folded into base
  const int baseB = nlo + (g1 ? 8 * 16 : 0);
  v8f acc = vzero8();
  sfor<7>([&](auto fi) {                            // K = 216 padded to 224
    constexpr int F = decltype(fi)::v;
    v16bf a;
    sfor<16>([&](auto ii) {
      constexpr int idx = decltype(ii)::v;
      constexpr int v = idx >> 1, p = idx & 1;
      constexpr int kc = F * 32 + (v & 3) * 2 + (v >> 2) * 16 + p;  // grp0 k
      constexpr bool vA = kc < 216, vB = (kc + 8) < 216;
      if constexpr (!vA && !vB) a[idx] = (__bf16)0.0f;
      else {
        const __bf16 t = x[baseA + kc];             // immediate offset kc
        if constexpr (vA && vB) a[idx] = t;
        else                    a[idx] = ((g1 ? vB : vA)) ? t : (__bf16)0.0f;
      }
    });
    v16bf b = make_b_frag<16, F * 32>(wl, baseB);
    acc = wmma_bf16(a, b, acc);
  });
  sfor<8>([&](auto ri) {
    constexpr int r = decltype(ri)::v;
    const int m = r + (g1 ? 8 : 0);
    if (nlo < 10) out[(size_t)(imgbase + m) * 10 + nlo] = acc[r];
  });
}

// ---- OIHW f32 weights -> bf16 im2col B [KPAD][NT*16] in LDS (zero padded).
template <int O, int C, int KH, int KW, int KPAD, int NT>
static __device__ __forceinline__ void load_conv_w(const float* g, __bf16* wl, int tid) {
  constexpr int NSTR = NT * 16, KTRUE = C * KH * KW;
  for (int i = tid; i < KPAD * NSTR; i += NTHREADS) {
    const int k = i / NSTR, n = i % NSTR;
    float v = 0.f;
    if (k < KTRUE && n < O) v = g[n * KTRUE + k];
    wl[i] = (__bf16)v;
  }
}

static __device__ __forceinline__ void zero_region(__bf16* p, int nelts, int tid) {
  unsigned int* q = (unsigned int*)p;              // regions are dword aligned/sized
  for (int i = tid; i < nelts / 2; i += NTHREADS) q[i] = 0u;
}

__global__ __launch_bounds__(NTHREADS) void mnist_fused(
    const float* __restrict__ inp, const float* __restrict__ w1,
    const float* __restrict__ w2, const float* __restrict__ w3,
    const float* __restrict__ w4, const float* __restrict__ w5,
    const float* __restrict__ wd, float* __restrict__ out) {
  __shared__ __bf16 smem[LDS_ELTS];
  const int tid = threadIdx.x;
  const int lane = tid & 31;
  const int wave = __builtin_amdgcn_readfirstlane(tid) >> 5;   // SGPR wave id
  const int imgbase = blockIdx.x * TILE;

  // P0: zero input + act1 regions (halos), stage all weights as bf16 im2col.
  zero_region(smem + R1_OFF, TILE * 1 * 34 * 34, tid);
  zero_region(smem + R0_OFF, TILE * 3 * 32 * 32, tid);
  load_conv_w<3, 1, 7, 7, 64, 1>(w1, smem + W1_OFF, tid);
  load_conv_w<8, 3, 5, 5, 96, 1>(w2, smem + W2_OFF, tid);
  load_conv_w<10, 8, 3, 3, 96, 1>(w3, smem + W3_OFF, tid);
  load_conv_w<16, 10, 3, 3, 96, 1>(w4, smem + W4_OFF, tid);
  load_conv_w<24, 16, 3, 3, 160, 2>(w5, smem + W5_OFF, tid);
  for (int i = tid; i < 224 * 16; i += NTHREADS) {  // dense: B[k][n] = wd[n][k]
    const int k = i / 16, n = i % 16;
    float v = 0.f;
    if (k < 216 && n < 10) v = wd[n * 216 + k];
    smem[WD_OFF + i] = (__bf16)v;
  }
  __syncthreads();

  // P1: input interior f32 -> bf16 (halo 3 for conv1's pad).
  {
    const float* src = inp + (size_t)imgbase * 784;
    for (int i = tid; i < TILE * 784; i += NTHREADS) {
      const int m = i / 784, r = i % 784, h = r / 28, w = r % 28;
      smem[R1_OFF + (m * 34 + h + 3) * 34 + w + 3] = (__bf16)src[i];
    }
  }
  __syncthreads();

  // conv1 7x7 pad3: R1 [16][1][34][34] -> R0 [16][3][32][32] (halo2 for conv2)
  conv_relu<1, 28, 28, 7, 7, 3, 3, 64, 2>(smem + R1_OFF, smem + R0_OFF, smem + W1_OFF,
                                          wave, lane);
  __syncthreads();
  // P3: zero actP1 (R1) and act3 (R2) regions for their halos.
  zero_region(smem + R1_OFF, TILE * 8 * 16 * 16, tid);
  zero_region(smem + R2_OFF, TILE * 10 * 16 * 16, tid);
  __syncthreads();
  // conv2 5x5 pad2 + pool(3,2,1): R0 -> R1 [16][8][16][16] (halo1)
  conv_relu_pool<3, 28, 28, 5, 5, 2, 8, 96, 14, 14, 1, 1, 1>(
      smem + R0_OFF, smem + R1_OFF, smem + W2_OFF, wave, lane);
  __syncthreads();
  // P5: zero actP2 (R0) region (halo1) — R0 is idle between conv2 and conv4.
  zero_region(smem + R0_OFF, TILE * 16 * 9 * 9, tid);
  __syncthreads();
  // conv3 3x3 pad1: R1 -> R2 [16][10][16][16] (halo1)
  conv_relu<8, 14, 14, 3, 3, 1, 10, 96, 1>(smem + R1_OFF, smem + R2_OFF, smem + W3_OFF,
                                           wave, lane);
  __syncthreads();
  // conv4 3x3 pad1 + pool(3,2,1): R2 -> R0 [16][16][9][9] (halo1)
  conv_relu_pool<10, 14, 14, 3, 3, 1, 16, 96, 7, 7, 1, 1, 1>(
      smem + R2_OFF, smem + R0_OFF, smem + W4_OFF, wave, lane);
  __syncthreads();
  // conv5 3x3 pad1 + pool(3,2,0): R0 -> R1 [16][24][3][3] (no halo; fully written)
  conv_relu_pool<16, 7, 7, 3, 3, 1, 24, 160, 3, 3, 0, 0, 2>(
      smem + R0_OFF, smem + R1_OFF, smem + W5_OFF, wave, lane);
  __syncthreads();
  // dense [16x216] @ [216x10] -> global f32
  if (wave == 0) dense_out(smem + R1_OFF, smem + WD_OFF, out, imgbase, lane);
}

extern "C" void kernel_launch(void* const* d_in, const int* in_sizes, int n_in,
                              void* d_out, int out_size, void* d_ws, size_t ws_size,
                              hipStream_t stream) {
  (void)n_in; (void)out_size; (void)d_ws; (void)ws_size;
  const float* inp = (const float*)d_in[0];
  const float* w1  = (const float*)d_in[1];
  const float* w2  = (const float*)d_in[2];
  const float* w3  = (const float*)d_in[3];
  const float* w4  = (const float*)d_in[4];
  const float* w5  = (const float*)d_in[5];
  const float* wd  = (const float*)d_in[6];
  float* out = (float*)d_out;

  const int B = in_sizes[0] / 784;        // 16384
  const int nblocks = B / TILE;           // 1024 workgroups of 8 wave32s
  mnist_fused<<<nblocks, NTHREADS, 0, stream>>>(inp, w1, w2, w3, w4, w5, wd, out);
}
